// GCNAggregator_72164040508244
// MI455X (gfx1250) — compile-verified, matrix operand order checked by reading, not run on hardware
//
#include <hip/hip_runtime.h>

typedef __attribute__((ext_vector_type(16))) __bf16 v16bf;
typedef __attribute__((ext_vector_type(8)))  __bf16 v8bf;
typedef __attribute__((ext_vector_type(8)))  float  v8f;
typedef __attribute__((ext_vector_type(4)))  float  v4f;

#define DIM 128

__device__ __forceinline__ void atomic_add_f32(float* p, float v) {
    __hip_atomic_fetch_add(p, v, __ATOMIC_RELAXED, __HIP_MEMORY_SCOPE_AGENT);
}

// K index pattern of 16-bit A/B WMMA fragments (CDNA5 ISA 7.12.2):
// element e (0..15) of lane holds K = (e<8 ? e : e+8) + (lane<16 ? 0 : 8)
__device__ __forceinline__ int kpat(int e, int lane) {
    return ((e < 8) ? e : (e + 8)) + ((lane & 16) ? 8 : 0);
}

// ---------------- zero workspace (agg [N*128] followed contiguously by deg [N]) ----
__global__ void zero_f32(float* p, int n) {
    int i = blockIdx.x * 256 + threadIdx.x;
    if (i < n) p[i] = 0.0f;
}

// ---------------- split W into bf16 hi/lo, pre-swizzled to B-fragment layout -------
// slot t = ((kstep*8 + wtile)*32 + lane)*16 + e
__global__ void w_split_swizzle(const float* __restrict__ W,
                                __bf16* __restrict__ whi, __bf16* __restrict__ wlo) {
    int t = blockIdx.x * 256 + threadIdx.x;       // 0..16383
    int e     = t & 15;
    int lane  = (t >> 4) & 31;
    int wtile = (t >> 9) & 7;
    int kstep = t >> 12;
    int k = kstep * 32 + kpat(e, lane);
    int n = wtile * 16 + (lane & 15);
    float v = W[k * DIM + n];
    __bf16 hi = (__bf16)v;
    __bf16 lo = (__bf16)(v - (float)hi);
    whi[t] = hi;
    wlo[t] = lo;
}

// ---------------- edge scatter: agg[row] += neigh[col]; deg[row] += 1 --------------
// 32 threads per edge, 4 floats each (float4 load, 4 f32 atomics -> contiguous 512B/edge)
__global__ void edge_scatter(const float* __restrict__ neigh,
                             const int* __restrict__ arow, const int* __restrict__ acol,
                             float* __restrict__ agg, float* __restrict__ deg,
                             long long total) {
    long long t = (long long)blockIdx.x * 256 + threadIdx.x;
    if (t >= total) return;
    int e = (int)(t >> 5);
    int q = (int)(t & 31);
    int r = arow[e];
    int c = acol[e];
    v4f nv = *(const v4f*)(neigh + (long long)c * DIM + q * 4);
    float* dst = agg + (long long)r * DIM + q * 4;
    atomic_add_f32(dst + 0, nv.x);
    atomic_add_f32(dst + 1, nv.y);
    atomic_add_f32(dst + 2, nv.z);
    atomic_add_f32(dst + 3, nv.w);
    if (q == 0) atomic_add_f32(deg + r, 1.0f);
}

// ---------------- fused: x = self + agg/max(deg,1e-7); out = relu(x @ W) -----------
// 1 block = 16 output rows; 8 waves; wave w computes 16x16 tile at columns [16w,16w+16)
// bf16x3: acc += hi*hi + hi*lo + lo*hi  (~fp32 accuracy, fp32 accumulate)
__global__ __launch_bounds__(256) void gcn_gemm(const float* __restrict__ selfe,
                                                const float* __restrict__ agg,
                                                const float* __restrict__ deg,
                                                const __bf16* __restrict__ whi,
                                                const __bf16* __restrict__ wlo,
                                                float* __restrict__ out) {
    __shared__ alignas(32) __bf16 xhi[16 * DIM];
    __shared__ alignas(32) __bf16 xlo[16 * DIM];

    const int tid  = threadIdx.x;
    const int row0 = blockIdx.x * 16;

    // Build normalized x tile in LDS as bf16 hi/lo (each thread: 8 elements)
#pragma unroll
    for (int i = 0; i < 8; ++i) {
        int f = tid + 256 * i;            // 0..2047 over [16 rows][128 cols]
        int m = f >> 7;
        int d = f & 127;
        long long gr = row0 + m;
        float dg = deg[gr];
        float xv = selfe[gr * DIM + d] + agg[gr * DIM + d] / fmaxf(dg, 1e-7f);
        __bf16 hi = (__bf16)xv;
        xhi[f] = hi;
        xlo[f] = (__bf16)(xv - (float)hi);
    }
    __syncthreads();

    const int lane = tid & 31;
    const int w    = tid >> 5;            // column tile 0..7
    const int m    = lane & 15;
    const int off  = (lane & 16) ? 8 : 0; // lane-half K offset

    const v8bf* rhi = (const v8bf*)(xhi + m * DIM);
    const v8bf* rlo = (const v8bf*)(xlo + m * DIM);

    v8f acc = {};
#pragma unroll
    for (int ks = 0; ks < 4; ++ks) {
        const int k0 = ks * 32;
        union Frag { v16bf v; v8bf h[2]; } ah, al, bh, bl;
        // A fragment: two contiguous 8-element (16B) runs matching kpat()
        ah.h[0] = rhi[(k0 + off) >> 3];
        ah.h[1] = rhi[(k0 + 16 + off) >> 3];
        al.h[0] = rlo[(k0 + off) >> 3];
        al.h[1] = rlo[(k0 + 16 + off) >> 3];
        // B fragment: pre-swizzled, contiguous 32B per lane
        long long bidx = ((((long long)ks * 8 + w) * 32) + lane) * 16;
        bh.v = *(const v16bf*)(whi + bidx);
        bl.v = *(const v16bf*)(wlo + bidx);

        acc = __builtin_amdgcn_wmma_f32_16x16x32_bf16(false, ah.v, false, bh.v,
                                                      (short)0, acc, false, false);
        acc = __builtin_amdgcn_wmma_f32_16x16x32_bf16(false, ah.v, false, bl.v,
                                                      (short)0, acc, false, false);
        acc = __builtin_amdgcn_wmma_f32_16x16x32_bf16(false, al.v, false, bh.v,
                                                      (short)0, acc, false, false);
    }

    // D layout: VGPR r -> M = r + (lane<16 ? 0 : 8), N = lane&15
    const int col0 = w * 16;
    const int n    = lane & 15;
    const int mb   = (lane & 16) ? 8 : 0;
#pragma unroll
    for (int r = 0; r < 8; ++r) {
        out[(long long)(row0 + mb + r) * DIM + col0 + n] = fmaxf(acc[r], 0.0f);
    }
}

extern "C" void kernel_launch(void* const* d_in, const int* in_sizes, int n_in,
                              void* d_out, int out_size, void* d_ws, size_t ws_size,
                              hipStream_t stream) {
    const float* selfe = (const float*)d_in[0];
    const float* neigh = (const float*)d_in[1];
    const float* W     = (const float*)d_in[2];
    const int*   arow  = (const int*)d_in[3];
    const int*   acol  = (const int*)d_in[4];
    float* out = (float*)d_out;

    const int N = in_sizes[0] / DIM;   // 100000
    const int E = in_sizes[3];         // 1600000

    char* ws = (char*)d_ws;
    size_t off = 0;
    float* agg = (float*)(ws + off); off += (size_t)N * DIM * sizeof(float);
    float* deg = (float*)(ws + off); off += (size_t)N * sizeof(float);
    off = (off + 255) & ~(size_t)255;
    __bf16* whi = (__bf16*)(ws + off); off += (size_t)16384 * sizeof(__bf16);
    off = (off + 255) & ~(size_t)255;
    __bf16* wlo = (__bf16*)(ws + off);

    // 1) zero agg + deg (deg is contiguous after agg)
    int nz = N * DIM + N;
    zero_f32<<<(nz + 255) / 256, 256, 0, stream>>>(agg, nz);

    // 2) split + swizzle W into bf16 hi/lo B-fragments
    w_split_swizzle<<<16384 / 256, 256, 0, stream>>>(W, whi, wlo);

    // 3) edge scatter (L2-resident atomics: neigh is 51MB, fits 192MB L2)
    long long total = (long long)E * 32;
    long long blocks = (total + 255) / 256;
    edge_scatter<<<(unsigned int)blocks, 256, 0, stream>>>(neigh, arow, acol, agg, deg, total);

    // 4) fused normalize + bf16x3 WMMA GEMM + relu
    gcn_gemm<<<N / 16, 256, 0, stream>>>(selfe, agg, deg, whi, wlo, out);
}
